// GraphAttentionConvLayer_52140902974081
// MI455X (gfx1250) — compile-verified, matrix operand order checked by reading, not run on hardware
//
#include <hip/hip_runtime.h>
#include <hip/hip_bf16.h>
#include <math.h>

typedef __attribute__((ext_vector_type(16))) _Float16 v16h;
typedef __attribute__((ext_vector_type(8)))  float    v8f;
typedef __attribute__((ext_vector_type(4)))  int      v4i;

#define AS1 __attribute__((address_space(1)))
#define AS3 __attribute__((address_space(3)))

// Problem constants
#define BB 16
#define NN 4096
#define PP 1024
#define SS 32
#define CC 64
#define DD 128          // final feature dim
#define ROWS 128        // 4 points * 32 neighbors per workgroup
#define KP1 96          // padded K for layer1: [xyz 0..2 | zeros 3..7 | feat 8..71 | zeros]
#define FCOL 8          // feature start column (16B aligned within a row)
#define KPA 160         // padded K for attention (131 -> 160)

// Workspace byte offsets (32B aligned)
#define OFF_PTS   ((size_t)0)                   // ptsT f16: 16*4096*64 halfs = 8388608 B
#define OFF_W1    ((size_t)8388608)             // 6144 halfs  (Kt=3, Nt=4)
#define OFF_W2    ((size_t)8400896)             // 4096 halfs  (Kt=2, Nt=4)
#define OFF_W3    ((size_t)8409088)             // 8192 halfs  (Kt=2, Nt=8)
#define OFF_A     ((size_t)8425472)             // 20480 halfs (Kt=5, Nt=8)
#define OFF_SCSH  ((size_t)8466432)             // 512 floats

// ---------------- prep: transpose points (B,C,N) -> (B,N,C) f16 ----------------
__global__ void prep_transpose(const float* __restrict__ points, _Float16* __restrict__ ptsT) {
    unsigned i = blockIdx.x * 256u + threadIdx.x;           // total 16*64*4096 = 4194304
    if (i >= (unsigned)(BB * CC * NN)) return;
    unsigned b = i >> 18;                                   // CC*NN = 262144 = 2^18
    unsigned r = i & 262143u;
    unsigned c = r >> 12;                                   // NN = 4096 = 2^12
    unsigned n = r & 4095u;
    ptsT[((size_t)(b * NN + n)) * CC + c] = (_Float16)points[i];
}

// ---------------- prep: pack W^T into B-fragment order ----------------
// packed element (kt,nt,lane,j): k = kt*32 + 16*(v/4) + 2*(v%4) + 8*(lane/16) + (j&1),
// n = nt*16 + lane%16, v = j/2.  Each lane's 16 halfs are contiguous (32B).
// mode 0: W is (co,ci) row-major.  mode 1: a-matrix (K,N) row-major.
// mode 2: layer1 with shifted layout (k<3 -> xyz weights, 8<=k<72 -> feat weight k-5).
__device__ __forceinline__ void pack_one(int idx, int Nt, int ci, int co,
                                         const float* __restrict__ W, int mode,
                                         _Float16* __restrict__ dst) {
    int j    = idx & 15;
    int lane = (idx >> 4) & 31;
    int rest = idx >> 9;
    int nt   = rest % Nt;
    int kt   = rest / Nt;
    int g    = lane >> 4;
    int n    = nt * 16 + (lane & 15);
    int v    = j >> 1;
    int k    = kt * 32 + 16 * (v >> 2) + 2 * (v & 3) + 8 * g + (j & 1);
    float val = 0.0f;
    if (n < co) {
        if (mode == 0) {
            if (k < ci) val = W[n * ci + k];
        } else if (mode == 1) {
            if (k < ci) val = W[k * co + n];
        } else { // layer1 shifted
            if (k < 3)                           val = W[n * 67 + k];
            else if (k >= FCOL && k < FCOL + 64) val = W[n * 67 + (k - FCOL + 3)];
        }
    }
    dst[idx] = (_Float16)val;
}

__global__ void prep_pack(const float* __restrict__ w1, const float* __restrict__ w2,
                          const float* __restrict__ w3, const float* __restrict__ a,
                          _Float16* __restrict__ pW1, _Float16* __restrict__ pW2,
                          _Float16* __restrict__ pW3, _Float16* __restrict__ pA) {
    int i = blockIdx.x * 256 + threadIdx.x;
    const int s1 = 3 * 4 * 32 * 16;   // 6144
    const int s2 = 2 * 4 * 32 * 16;   // 4096
    const int s3 = 2 * 8 * 32 * 16;   // 8192
    const int sa = 5 * 8 * 32 * 16;   // 20480
    if (i < s1)                 { pack_one(i,            4, 67,  64, w1, 2, pW1); return; }
    if (i < s1 + s2)            { pack_one(i - s1,       4, 64,  64, w2, 0, pW2); return; }
    if (i < s1 + s2 + s3)       { pack_one(i - s1 - s2,  8, 64, 128, w3, 0, pW3); return; }
    if (i < s1 + s2 + s3 + sa)  { pack_one(i - s1 - s2 - s3, 8, 131, 128, a, 1, pA); return; }
}

// ---------------- prep: fold bias + BN into per-channel scale/shift ----------------
__global__ void prep_scsh(const float* b1, const float* g1, const float* be1, const float* m1, const float* v1,
                          const float* b2, const float* g2, const float* be2, const float* m2, const float* v2,
                          const float* b3, const float* g3, const float* be3, const float* m3, const float* v3,
                          float* __restrict__ scsh) {
    int t = threadIdx.x;   // 256 threads
    const float eps = 1e-5f;
    if (t < 64) {
        float sc = g1[t] * rsqrtf(v1[t] + eps);
        scsh[t]       = sc;
        scsh[64 + t]  = (b1[t] - m1[t]) * sc + be1[t];
    } else if (t < 128) {
        int c = t - 64;
        float sc = g2[c] * rsqrtf(v2[c] + eps);
        scsh[128 + c] = sc;
        scsh[192 + c] = (b2[c] - m2[c]) * sc + be2[c];
    } else {
        int c = t - 128;   // 0..127
        float sc = g3[c] * rsqrtf(v3[c] + eps);
        scsh[256 + c] = sc;
        scsh[384 + c] = (b3[c] - m3[c]) * sc + be3[c];
    }
}

// ---------------- fragment loaders ----------------
__device__ __forceinline__ v16h load_a_frag(const _Float16* __restrict__ lds, int ldin,
                                            int mt, int kt, int lane) {
    int g  = lane >> 4;
    int mr = lane & 15;
    const _Float16* p = lds + (mt * 16 + mr) * ldin + kt * 32 + 8 * g;
    v16h a;
#pragma unroll
    for (int v = 0; v < 8; ++v) {
        int kb = 16 * (v >> 2) + 2 * (v & 3);
        a[2 * v]     = p[kb];
        a[2 * v + 1] = p[kb + 1];
    }
    return a;
}

// ---------------- mini-GEMM (M=128): A frags hoisted, fully unrolled ----------------
template <int Kt, int Nt>
__device__ __forceinline__ void gemm_bn(const _Float16* __restrict__ lin, int ldin,
                                        const _Float16* __restrict__ wfrag,
                                        const float* __restrict__ sc, const float* __restrict__ sh,
                                        _Float16* __restrict__ lout, int ldout) {
    const int tid  = threadIdx.x;
    const int wave = tid >> 5;
    const int lane = tid & 31;
    const int g    = lane >> 4;
    const int nr   = lane & 15;
    const v16h* wf = (const v16h*)wfrag;
    v16h a0[Kt], a1[Kt];
#pragma unroll
    for (int kt = 0; kt < Kt; ++kt) {
        a0[kt] = load_a_frag(lin, ldin, wave * 2,     kt, lane);
        a1[kt] = load_a_frag(lin, ldin, wave * 2 + 1, kt, lane);
    }
#pragma unroll
    for (int nt = 0; nt < Nt; ++nt) {
        v8f c0 = {}; v8f c1 = {};
#pragma unroll
        for (int kt = 0; kt < Kt; ++kt) {
            v16h bm = wf[(kt * Nt + nt) * 32 + lane];
            c0 = __builtin_amdgcn_wmma_f32_16x16x32_f16(false, a0[kt], false, bm, (short)0, c0, false, false);
            c1 = __builtin_amdgcn_wmma_f32_16x16x32_f16(false, a1[kt], false, bm, (short)0, c1, false, false);
        }
        int n = nt * 16 + nr;
        float scn = sc[n], shn = sh[n];
        int r0 = wave * 32 + 8 * g;
        int r1 = r0 + 16;
#pragma unroll
        for (int r = 0; r < 8; ++r) {
            float h0 = fmaxf(c0[r] * scn + shn, 0.0f);
            float h1 = fmaxf(c1[r] * scn + shn, 0.0f);
            lout[(r0 + r) * ldout + n] = (_Float16)h0;
            lout[(r1 + r) * ldout + n] = (_Float16)h1;
        }
    }
}

template <int Kt, int Nt>
__device__ __forceinline__ void gemm_leaky(const _Float16* __restrict__ lin, int ldin,
                                           const _Float16* __restrict__ wfrag,
                                           float* __restrict__ eout, int ldout) {
    const int tid  = threadIdx.x;
    const int wave = tid >> 5;
    const int lane = tid & 31;
    const int g    = lane >> 4;
    const int nr   = lane & 15;
    const v16h* wf = (const v16h*)wfrag;
    v16h a0[Kt], a1[Kt];
#pragma unroll
    for (int kt = 0; kt < Kt; ++kt) {
        a0[kt] = load_a_frag(lin, ldin, wave * 2,     kt, lane);
        a1[kt] = load_a_frag(lin, ldin, wave * 2 + 1, kt, lane);
    }
#pragma unroll
    for (int nt = 0; nt < Nt; ++nt) {
        v8f c0 = {}; v8f c1 = {};
#pragma unroll
        for (int kt = 0; kt < Kt; ++kt) {
            v16h bm = wf[(kt * Nt + nt) * 32 + lane];
            c0 = __builtin_amdgcn_wmma_f32_16x16x32_f16(false, a0[kt], false, bm, (short)0, c0, false, false);
            c1 = __builtin_amdgcn_wmma_f32_16x16x32_f16(false, a1[kt], false, bm, (short)0, c1, false, false);
        }
        int n  = nt * 16 + nr;
        int r0 = wave * 32 + 8 * g;
        int r1 = r0 + 16;
#pragma unroll
        for (int r = 0; r < 8; ++r) {
            float e0 = c0[r]; e0 = (e0 > 0.0f) ? e0 : 0.2f * e0;
            float e1 = c1[r]; e1 = (e1 > 0.0f) ? e1 : 0.2f * e1;
            eout[(r0 + r) * ldout + n] = e0;
            eout[(r1 + r) * ldout + n] = e1;
        }
    }
}

// ---------------- tiny dense layer for the 4 fps rows (VALU) ----------------
__device__ __forceinline__ void fps_dense(const float* __restrict__ fin, float* __restrict__ fout,
                                          int ci, int co, const float* __restrict__ W,
                                          const float* __restrict__ sc, const float* __restrict__ sh) {
    int tid = threadIdx.x;
    for (int task = tid; task < 4 * co; task += 128) {
        int pt = task / co;
        int o  = task - pt * co;
        const float* xr = fin + pt * 128;
        const float* wr = W + o * ci;
        float acc = 0.0f;
        for (int k = 0; k < ci; ++k) acc += xr[k] * wr[k];
        fout[pt * 128 + o] = fmaxf(acc * sc[o] + sh[o], 0.0f);
    }
}

// ================= fused main kernel: 1 WG (128 threads) per 4 points =================
__global__ void __launch_bounds__(128)
fused_sa_attn(const float* __restrict__ xyz, const int* __restrict__ fps_idx,
              const int* __restrict__ group_idx, const _Float16* __restrict__ ptsT,
              const float* __restrict__ w1raw, const float* __restrict__ w2raw,
              const float* __restrict__ w3raw,
              const _Float16* __restrict__ pW1, const _Float16* __restrict__ pW2,
              const _Float16* __restrict__ pW3, const _Float16* __restrict__ pA,
              const float* __restrict__ scsh, float* __restrict__ out) {
    __shared__ _Float16 sBuf0[ROWS * KPA];    // 40 KB : X0 / H2 / Acat  (row = 320B, 16B-aligned)
    __shared__ _Float16 sBuf1[ROWS * DD];     // 32 KB : H1 / H3
    __shared__ float    sE[ROWS * DD];        // 64 KB : attention scores
    __shared__ float    sRel[ROWS * 3];
    __shared__ float    sNew[4 * 3];
    __shared__ int      sFpsN[4];
    __shared__ int      sNIdx[ROWS];
    __shared__ float    sFpsA[4 * 128];
    __shared__ float    sFpsB[4 * 128];

    const int tid  = threadIdx.x;
    const int b    = blockIdx.x >> 8;         // P/4 = 256 blocks per batch
    const int p0   = (blockIdx.x & 255) * 4;

    const float* sc1 = scsh;       const float* sh1 = scsh + 64;
    const float* sc2 = scsh + 128; const float* sh2 = scsh + 192;
    const float* sc3 = scsh + 256; const float* sh3 = scsh + 384;

    // prefetch attention weights (largest reused block) into cache hierarchy
    __builtin_prefetch(pA + (size_t)tid * 160, 0, 0);

    // ---- stage 0: indices + centroids ----
    if (tid < 4) sFpsN[tid] = fps_idx[b * PP + p0 + tid];
    __syncthreads();
    if (tid < 12) {
        int pt = tid / 3, d = tid % 3;
        sNew[tid] = xyz[(b * 3 + d) * NN + sFpsN[pt]];
    }
    sNIdx[tid] = group_idx[(b * PP + p0 + (tid >> 5)) * SS + (tid & 31)];
    __syncthreads();

    // ---- stage 1: gather X0 = [rel_xyz | 0 | feat] into sBuf0 (ld=KPA) ----
    {
        // feature block: 128B contiguous per neighbor row -> async VMEM->LDS copy
        int n = sNIdx[tid];
        const _Float16* gsrc = ptsT + ((size_t)(b * NN + n)) * CC;
        _Float16* ldst = &sBuf0[tid * KPA + FCOL];
#if __has_builtin(__builtin_amdgcn_global_load_async_to_lds_b128)
#pragma unroll
        for (int ch = 0; ch < 8; ++ch) {
            AS1 v4i* gp = (AS1 v4i*)(uintptr_t)(gsrc + ch * 8);
            AS3 v4i* lp = (AS3 v4i*)(unsigned)(uintptr_t)(ldst + ch * 8);
            __builtin_amdgcn_global_load_async_to_lds_b128(gp, lp, 0, 0);
        }
#else
        struct alignas(16) H8 { _Float16 h[8]; };
#pragma unroll
        for (int ch = 0; ch < 8; ++ch)
            *(H8*)(ldst + ch * 8) = *(const H8*)(gsrc + ch * 8);
#endif
        // xyz (relative) + zero padding cols 3..7
        for (int e = tid; e < ROWS * FCOL; e += 128) {
            int row = e >> 3, c = e & 7;
            _Float16 hv;
            if (c < 3) {
                float rel = xyz[(b * 3 + c) * NN + sNIdx[row]] - sNew[(row >> 5) * 3 + c];
                sRel[row * 3 + c] = rel;
                hv = (_Float16)rel;
            } else {
                hv = (_Float16)0.0f;
            }
            sBuf0[row * KPA + c] = hv;
        }
        // zero padding cols 72..95
        for (int e = tid; e < ROWS * (KP1 - FCOL - CC); e += 128) {
            int row = e / 24, c = FCOL + CC + (e % 24);
            sBuf0[row * KPA + c] = (_Float16)0.0f;
        }
    }
    for (int e = tid; e < 4 * 67; e += 128) {
        int pt = e / 67, c = e - pt * 67;
        int n = sFpsN[pt];
        float v = (c < 3) ? sNew[pt * 3 + c] : (float)ptsT[((size_t)(b * NN + n)) * CC + (c - 3)];
        sFpsA[pt * 128 + c] = v;
    }
#if __has_builtin(__builtin_amdgcn_global_load_async_to_lds_b128)
#if __has_builtin(__builtin_amdgcn_s_wait_asynccnt)
    __builtin_amdgcn_s_wait_asynccnt(0);
#else
    asm volatile("s_wait_asynccnt 0" ::: "memory");
#endif
#endif
    __syncthreads();

    // ---- stage 2: layer1 (K=96 -> 64) ----
    gemm_bn<3, 4>(sBuf0, KPA, pW1, sc1, sh1, sBuf1, DD);
    fps_dense(sFpsA, sFpsB, 67, 64, w1raw, sc1, sh1);
    __syncthreads();

    // ---- stage 3: layer2 (64 -> 64) ----
    gemm_bn<2, 4>(sBuf1, DD, pW2, sc2, sh2, sBuf0, KPA);
    fps_dense(sFpsB, sFpsA, 64, 64, w2raw, sc2, sh2);
    __syncthreads();

    // ---- stage 4: layer3 (64 -> 128) ----
    gemm_bn<2, 8>(sBuf0, KPA, pW3, sc3, sh3, sBuf1, DD);
    fps_dense(sFpsA, sFpsB, 64, 128, w3raw, sc3, sh3);
    __syncthreads();

    // ---- stage 5: build Acat = [delta_p(3) | delta_h(128)] padded to 160 ----
    for (int e = tid; e < ROWS * KPA; e += 128) {
        int row = e / KPA, c = e - row * KPA;
        int pt = row >> 5;
        _Float16 hv;
        if (c < 3) {
            hv = (_Float16)(sNew[pt * 3 + c] - sRel[row * 3 + c]);
        } else if (c < 131) {
            int d = c - 3;
            hv = (_Float16)(sFpsB[pt * 128 + d] - (float)sBuf1[row * DD + d]);
        } else {
            hv = (_Float16)0.0f;
        }
        sBuf0[e] = hv;
    }
    __syncthreads();

    // ---- stage 6: attention scores e = leaky(Acat @ a) ----
    gemm_leaky<5, 8>(sBuf0, KPA, pA, sE, DD);
    __syncthreads();

    // ---- stage 7: softmax over S and attention-weighted pool ----
    float* out2 = out + BB * 3 * PP;
    for (int task = tid; task < 4 * DD; task += 128) {
        int pt = task >> 7, d = task & 127;
        int base = (pt * 32) * DD + d;
        float mx = -INFINITY;
#pragma unroll
        for (int s = 0; s < SS; ++s) mx = fmaxf(mx, sE[base + s * DD]);
        float sum = 0.0f, acc = 0.0f;
#pragma unroll
        for (int s = 0; s < SS; ++s) {
            float w = __expf(sE[base + s * DD] - mx);
            sum += w;
            acc += w * (float)sBuf1[base + s * DD];
        }
        out2[((size_t)b * DD + d) * PP + p0 + pt] = acc / sum;
    }

    // ---- stage 8: new_xyz output (B,3,P) ----
    if (tid < 12) {
        int pt = tid / 3, d = tid % 3;
        out[(b * 3 + d) * PP + p0 + pt] = sNew[tid];
    }
}

extern "C" void kernel_launch(void* const* d_in, const int* in_sizes, int n_in,
                              void* d_out, int out_size, void* d_ws, size_t ws_size,
                              hipStream_t stream) {
    const float* xyz    = (const float*)d_in[0];
    const float* points = (const float*)d_in[1];
    const int*   fpsi   = (const int*)d_in[2];
    const int*   gidx   = (const int*)d_in[3];
    const float* w1 = (const float*)d_in[4];
    const float* b1 = (const float*)d_in[5];
    const float* g1 = (const float*)d_in[6];
    const float* be1 = (const float*)d_in[7];
    const float* m1 = (const float*)d_in[8];
    const float* v1 = (const float*)d_in[9];
    const float* w2 = (const float*)d_in[10];
    const float* b2 = (const float*)d_in[11];
    const float* g2 = (const float*)d_in[12];
    const float* be2 = (const float*)d_in[13];
    const float* m2 = (const float*)d_in[14];
    const float* v2 = (const float*)d_in[15];
    const float* w3 = (const float*)d_in[16];
    const float* b3 = (const float*)d_in[17];
    const float* g3 = (const float*)d_in[18];
    const float* be3 = (const float*)d_in[19];
    const float* m3 = (const float*)d_in[20];
    const float* v3 = (const float*)d_in[21];
    const float* a  = (const float*)d_in[22];

    char* ws = (char*)d_ws;
    _Float16* ptsT = (_Float16*)(ws + OFF_PTS);
    _Float16* pW1  = (_Float16*)(ws + OFF_W1);
    _Float16* pW2  = (_Float16*)(ws + OFF_W2);
    _Float16* pW3  = (_Float16*)(ws + OFF_W3);
    _Float16* pA   = (_Float16*)(ws + OFF_A);
    float*    scsh = (float*)(ws + OFF_SCSH);

    prep_transpose<<<(BB * CC * NN + 255) / 256, 256, 0, stream>>>(points, ptsT);
    prep_pack<<<(38912 + 255) / 256, 256, 0, stream>>>(w1, w2, w3, a, pW1, pW2, pW3, pA);
    prep_scsh<<<1, 256, 0, stream>>>(b1, g1, be1, m1, v1,
                                     b2, g2, be2, m2, v2,
                                     b3, g3, be3, m3, v3, scsh);
    fused_sa_attn<<<BB * (PP / 4), 128, 0, stream>>>(
        xyz, fpsi, gidx, ptsT, w1, w2, w3, pW1, pW2, pW3, pA, scsh, (float*)d_out);
}
